// FullyNestedTitans_34041910788570
// MI455X (gfx1250) — matrix operand order repaired and checked by reading, not compiled, not run on hardware
//
#include <hip/hip_runtime.h>
#include <math.h>

typedef __attribute__((ext_vector_type(16))) _Float16 v16h;
typedef __attribute__((ext_vector_type(8)))  _Float16 v8h;
typedef __attribute__((ext_vector_type(8)))  float    v8f;

#define BB   8
#define SS   512
#define HH   12
#define DD   64
#define DIMM 768
#define MM   (BB*SS)       // 4096 rows
#define SPAD 65            // LDS row pad (banks=64)

__device__ __forceinline__ float sigm(float x){ return 1.0f/(1.0f+__expf(-x)); }

// ---------------------------------------------------------------------------
// Kernel 1: input LayerNorm over DIM=768, emit f16 activations for WMMA GEMMs
// ---------------------------------------------------------------------------
__global__ __launch_bounds__(256) void ln_in_kernel(const float* __restrict__ x,
                                                    const float* __restrict__ w,
                                                    const float* __restrict__ b,
                                                    _Float16* __restrict__ xh)
{
    int m = blockIdx.x, tid = threadIdx.x;
    __shared__ float red[256];
    const float* row = x + (long)m*DIMM;
    float vals[3]; float s = 0.f;
    #pragma unroll
    for (int i=0;i<3;i++){ vals[i]=row[tid+i*256]; s+=vals[i]; }
    red[tid]=s; __syncthreads();
    for (int off=128; off; off>>=1){ if (tid<off) red[tid]+=red[tid+off]; __syncthreads(); }
    float mean = red[0]*(1.0f/DIMM); __syncthreads();
    float sq=0.f;
    #pragma unroll
    for (int i=0;i<3;i++){ float d=vals[i]-mean; sq+=d*d; }
    red[tid]=sq; __syncthreads();
    for (int off=128; off; off>>=1){ if (tid<off) red[tid]+=red[tid+off]; __syncthreads(); }
    float rs = rsqrtf(red[0]*(1.0f/DIMM) + 1e-5f);
    #pragma unroll
    for (int i=0;i<3;i++){
        int c = tid + i*256;
        xh[(long)m*DIMM + c] = (_Float16)((vals[i]-mean)*rs*w[c] + b[c]);
    }
}

// ---------------------------------------------------------------------------
// Kernel 2: convert weight f32 [K,N] -> f16 transposed [N,K] (contiguous-K B frags)
// ---------------------------------------------------------------------------
__global__ __launch_bounds__(256) void wconv_kernel(const float* __restrict__ W,
                                                    _Float16* __restrict__ Wt)
{
    int idx = blockIdx.x*256 + threadIdx.x;   // idx over N*K
    int n = idx / DIMM, k = idx % DIMM;
    Wt[idx] = (_Float16)W[(long)k*DIMM + n];
}

// ---------------------------------------------------------------------------
// WMMA GEMM, 2x4 wave blocking: each wave computes a 32x64 tile of
// C[4096x768] = A[4096x768] * Bt^T  (K=768).  A row-major [M,K] f16,
// Bt [N,K] f16.  Per K-step: 2 A frags reused across 4 B frags -> 8 WMMAs
// per 12 b128 loads.
// ---------------------------------------------------------------------------
#define NT (DIMM/64)            // 12 column blocks of 64
#define GEMM_BLOCKS ((MM/32)*NT/8)   // 192

__device__ __forceinline__ v16h load_fragA(const _Float16* p, int hi)
{
    v8h a0 = *(const v8h*)(p + hi*8);
    v8h a1 = *(const v8h*)(p + hi*8 + 16);
    v16h av;
    #pragma unroll
    for (int i=0;i<8;i++){ av[i]=a0[i]; av[8+i]=a1[i]; }
    return av;
}

__device__ __forceinline__ v16h load_fragB(const _Float16* p, int hi)
{
    v8h b0 = *(const v8h*)(p + hi*8);
    v8h b1 = *(const v8h*)(p + hi*8 + 16);
    v16h bv;
    #pragma unroll
    for (int i=0;i<8;i++){ bv[i]=b0[i]; bv[8+i]=b1[i]; }
    return bv;
}

// computes acc[2][4]; tm indexes 32-row blocks, tn indexes 64-col blocks
__device__ __forceinline__ void wmma_tile_2x4(const _Float16* __restrict__ A,
                                              const _Float16* __restrict__ Bt,
                                              int tm, int tn, int hi, int l15,
                                              v8f acc[2][4])
{
    const _Float16* arow0 = A  + (long)(tm*32 + l15)*DIMM;
    const _Float16* arow1 = arow0 + 16l*DIMM;
    const _Float16* brow0 = Bt + (long)(tn*64 + l15)*DIMM;

    for (int kb = 0; kb < DIMM; kb += 32) {
        v16h a0 = load_fragA(arow0 + kb, hi);
        v16h a1 = load_fragA(arow1 + kb, hi);
        v16h bf[4];
        #pragma unroll
        for (int j=0;j<4;j++) bf[j] = load_fragB(brow0 + (long)j*16*DIMM + kb, hi);
        #pragma unroll
        for (int j=0;j<4;j++){
            acc[0][j] = __builtin_amdgcn_wmma_f32_16x16x32_f16(false, a0, false, bf[j],
                                                               (short)0, acc[0][j], false, false);
            acc[1][j] = __builtin_amdgcn_wmma_f32_16x16x32_f16(false, a1, false, bf[j],
                                                               (short)0, acc[1][j], false, false);
        }
    }
}

__global__ __launch_bounds__(256) void gemm16_kernel(const _Float16* __restrict__ A,
                                                     const _Float16* __restrict__ Bt,
                                                     float* __restrict__ C)
{
    int wave = threadIdx.x >> 5, lane = threadIdx.x & 31;
    int tile = blockIdx.x*8 + wave;          // over (M/32)*(N/64)
    int tm = tile / NT, tn = tile % NT;
    int hi = lane >> 4, l15 = lane & 15;
    v8f acc[2][4] = {};
    wmma_tile_2x4(A, Bt, tm, tn, hi, l15, acc);
    #pragma unroll
    for (int a=0;a<2;a++){
        int crow = tm*32 + a*16 + hi*8;
        #pragma unroll
        for (int j=0;j<4;j++){
            int ccol = tn*64 + j*16 + l15;
            #pragma unroll
            for (int r=0;r<8;r++) C[(long)(crow+r)*DIMM + ccol] = acc[a][j][r];
        }
    }
}

// Final GEMM with fused sigmoid-gate epilogue: out = sigmoid(graw+bg) * (A@Wo)
__global__ __launch_bounds__(256) void gemm_out_kernel(const _Float16* __restrict__ A,
                                                       const _Float16* __restrict__ Bt,
                                                       const float* __restrict__ graw,
                                                       const float* __restrict__ bg,
                                                       float* __restrict__ C)
{
    int wave = threadIdx.x >> 5, lane = threadIdx.x & 31;
    int tile = blockIdx.x*8 + wave;
    int tm = tile / NT, tn = tile % NT;
    int hi = lane >> 4, l15 = lane & 15;
    v8f acc[2][4] = {};
    wmma_tile_2x4(A, Bt, tm, tn, hi, l15, acc);
    #pragma unroll
    for (int a=0;a<2;a++){
        int crow = tm*32 + a*16 + hi*8;
        #pragma unroll
        for (int j=0;j<4;j++){
            int ccol = tn*64 + j*16 + l15;
            #pragma unroll
            for (int r=0;r<8;r++){
                long idx = (long)(crow+r)*DIMM + ccol;
                float g = sigm(graw[idx] + bg[ccol]);
                C[idx] = g * acc[a][j][r];
            }
        }
    }
}

// ---------------------------------------------------------------------------
// Kernel 4: per-head LayerNorm over D=64 for k/v/q, permute to time-major [S,B,H,D]
// ---------------------------------------------------------------------------
__global__ __launch_bounds__(64) void headln_kernel(const float* __restrict__ kraw,
                                                    const float* __restrict__ vraw,
                                                    const float* __restrict__ qraw,
                                                    float* __restrict__ kn,
                                                    float* __restrict__ vn,
                                                    float* __restrict__ qn,
                                                    const float* __restrict__ kw, const float* __restrict__ kb,
                                                    const float* __restrict__ vw, const float* __restrict__ vb,
                                                    const float* __restrict__ qw, const float* __restrict__ qb)
{
    int id = blockIdx.x;                       // 3*B*S*H rows
    int which = id / (BB*SS*HH); int rem = id % (BB*SS*HH);
    int b = rem / (SS*HH); int r2 = rem % (SS*HH);
    int s = r2 / HH; int h = r2 % HH;
    const float* src = which==0 ? kraw : (which==1 ? vraw : qraw);
    float*       dst = which==0 ? kn   : (which==1 ? vn   : qn);
    const float* w   = which==0 ? kw   : (which==1 ? vw   : qw);
    const float* bia = which==0 ? kb   : (which==1 ? vb   : qb);
    int d = threadIdx.x;
    float v = src[((long)(b*SS+s))*DIMM + h*DD + d];
    __shared__ float red[64];
    red[d]=v; __syncthreads();
    for (int off=32; off; off>>=1){ if (d<off) red[d]+=red[d+off]; __syncthreads(); }
    float mean = red[0]*(1.0f/DD); __syncthreads();
    float df = v-mean; red[d]=df*df; __syncthreads();
    for (int off=32; off; off>>=1){ if (d<off) red[d]+=red[d+off]; __syncthreads(); }
    float rs = rsqrtf(red[0]*(1.0f/DD) + 1e-5f);
    dst[((long)((s*BB+b)*HH+h))*DD + d] = df*rs*w[d] + bia[d];
}

// ---------------------------------------------------------------------------
// Kernel 5: sequential Titans scan. One block per (b,h); 3 x 64x64 f32 states
// resident in LDS (row-padded). 512 steps.
// ---------------------------------------------------------------------------
__global__ __launch_bounds__(256) void scan_kernel(
    const float* __restrict__ kn, const float* __restrict__ vn, const float* __restrict__ qn,
    const float* __restrict__ mW1, const float* __restrict__ mb1,
    const float* __restrict__ mW2, const float* __restrict__ mb2,
    const float* __restrict__ lrs, const float* __restrict__ fgs,
    _Float16* __restrict__ outs_h, float* __restrict__ states_out)
{
    __shared__ float S0[DD*SPAD], S1[DD*SPAD], S2[DD*SPAD];
    __shared__ float kv[DD], vv[DD], qv[DD], outv[DD], sur[DD];
    __shared__ float p0v[DD], p1v[DD], p2v[DD];
    __shared__ float red3[8][4], redA[8], redB[8];
    __shared__ float ssv[4], effl[3], efff[3], blr[3], bfg[3];

    int tid = threadIdx.x;
    int bh = blockIdx.x; int b = bh / HH; int h = bh % HH;
    int row = tid >> 2, cb = (tid & 3) * 16;
    int lane = tid & 31, wave = tid >> 5;

    #pragma unroll
    for (int i=0;i<16;i++){
        S0[row*SPAD+cb+i]=0.f; S1[row*SPAD+cb+i]=0.f; S2[row*SPAD+cb+i]=0.f;
    }
    if (tid < 3) {
        const float LRSc[3] = {0.1f, 0.05f, 0.025f};
        blr[tid] = sigm(lrs[tid*HH+h]) * 2.0f*LRSc[tid];
        bfg[tid] = sigm(fgs[tid*HH+h]) * 0.5f;
    }
    __syncthreads();

    for (int t=0; t<SS; t++) {
        // stage 1: load k/v/q for this step; prefetch next step
        if (tid < 192) {
            int which = tid >> 6, d = tid & 63;
            const float* src = which==0 ? kn : (which==1 ? vn : qn);
            long idx = ((long)(t*BB + b)*HH + h)*DD + d;
            float val = src[idx];
            float* dstl = which==0 ? kv : (which==1 ? vv : qv);
            dstl[d] = val;
            if (t+1 < SS) __builtin_prefetch(src + idx + (long)BB*HH*DD, 0, 0);
        }
        __syncthreads();

        // stage 2: four matvecs (S0.q, S0.k, S1.k, S2.k), 64 threads each
        {
            int grp = tid >> 6, r = tid & 63;
            const float* Sm = (grp<=1) ? S0 : (grp==2 ? S1 : S2);
            const float* vec = (grp==0) ? qv : kv;
            float acc = 0.f;
            #pragma unroll 16
            for (int c=0;c<64;c++) acc += Sm[r*SPAD+c]*vec[c];
            if      (grp==0) outv[r] = acc;
            else if (grp==1) { p0v[r] = acc; sur[r] = acc - vv[r]; }
            else if (grp==2) p1v[r] = acc;
            else             p2v[r] = acc;
        }
        __syncthreads();

        // stage 3: means of S0, S1 and surprise (wave32 shuffle reduce)
        {
            float a0=0.f, a1=0.f;
            #pragma unroll
            for (int i=0;i<16;i++){ a0+=S0[row*SPAD+cb+i]; a1+=S1[row*SPAD+cb+i]; }
            float a2 = (tid<64) ? sur[tid] : 0.f;
            for (int off=16; off; off>>=1){
                a0 += __shfl_xor(a0, off, 32);
                a1 += __shfl_xor(a1, off, 32);
                a2 += __shfl_xor(a2, off, 32);
            }
            if (lane==0){ red3[wave][0]=a0; red3[wave][1]=a1; red3[wave][2]=a2; }
            __syncthreads();
            if (tid==0){
                float s0=0.f,s1=0.f,s2=0.f;
                #pragma unroll
                for (int wv=0;wv<8;wv++){ s0+=red3[wv][0]; s1+=red3[wv][1]; s2+=red3[wv][2]; }
                ssv[0]=s0*(1.0f/4096.f); ssv[1]=s1*(1.0f/4096.f); ssv[2]=s2*(1.0f/64.f);
            }
            __syncthreads();
        }

        // stage 4: meta MLP (input is zero-padded past 2 entries -> 2 axpy + gelu + 2 dots)
        {
            float d0=0.f, d1=0.f;
            if (tid < 128) {
                int l = tid >> 6, j = tid & 63;
                float ss = ssv[l], sus = ssv[2];
                float hv = ss*mW1[(l*2*DD + 0)*DD + j] + sus*mW1[(l*2*DD + 1)*DD + j] + mb1[l*DD + j];
                hv = 0.5f*hv*(1.0f + erff(hv*0.70710678f));   // exact gelu
                d0 = hv*mW2[(l*DD + j)*2 + 0];
                d1 = hv*mW2[(l*DD + j)*2 + 1];
            }
            for (int off=16; off; off>>=1){
                d0 += __shfl_xor(d0, off, 32);
                d1 += __shfl_xor(d1, off, 32);
            }
            if (lane==0 && wave<4){ redA[wave]=d0; redB[wave]=d1; }
            __syncthreads();
            if (tid < 2){
                int l = tid;
                float c0 = sigm(redA[2*l]+redA[2*l+1] + mb2[l*2+0]);
                float c1 = sigm(redB[2*l]+redB[2*l+1] + mb2[l*2+1]);
                effl[l] = blr[l]*(0.5f + c0);
                efff[l] = bfg[l]*(0.5f + c1);
            } else if (tid==2){ effl[2]=blr[2]; efff[2]=bfg[2]; }
            __syncthreads();
        }

        // stage 5: rank-1 updates of all 3 states + emit out_t (pre-update S0 read)
        {
            float pr0=p0v[row], pr1=p1v[row], pr2=p2v[row], su=sur[row];
            float u0 = efff[0]*pr0 + effl[0]*su;
            float u1 = efff[1]*pr1 + effl[1]*su;
            float u2 = efff[2]*pr2 + effl[2]*su;
            #pragma unroll
            for (int i=0;i<16;i++){
                float kc = kv[cb+i];
                S0[row*SPAD+cb+i] -= u0*kc;
                S1[row*SPAD+cb+i] -= u1*kc;
                S2[row*SPAD+cb+i] -= u2*kc;
            }
            if (tid < 64)
                outs_h[((long)b*SS + t)*DIMM + h*DD + tid] = (_Float16)outv[tid];
        }
        __syncthreads();
    }

    // final states -> d_out[BSD..] as [L,B,H,D,D]
    for (int l=0;l<3;l++){
        const float* Sm = (l==0) ? S0 : (l==1 ? S1 : S2);
        long base = ((long)(l*BB + b)*HH + h)*(DD*DD);
        #pragma unroll
        for (int i=0;i<16;i++)
            states_out[base + row*DD + cb + i] = Sm[row*SPAD+cb+i];
    }
}

// ---------------------------------------------------------------------------
extern "C" void kernel_launch(void* const* d_in, const int* in_sizes, int n_in,
                              void* d_out, int out_size, void* d_ws, size_t ws_size,
                              hipStream_t stream)
{
    const float* x      = (const float*)d_in[0];
    const float* Wk     = (const float*)d_in[1];
    const float* Wv     = (const float*)d_in[2];
    const float* Wq     = (const float*)d_in[3];
    const float* Wo     = (const float*)d_in[4];
    const float* ln_in_w= (const float*)d_in[5];
    const float* ln_in_b= (const float*)d_in[6];
    const float* ln_k_w = (const float*)d_in[7];
    const float* ln_k_b = (const float*)d_in[8];
    const float* ln_v_w = (const float*)d_in[9];
    const float* ln_v_b = (const float*)d_in[10];
    const float* ln_q_w = (const float*)d_in[11];
    const float* ln_q_b = (const float*)d_in[12];
    const float* mW1    = (const float*)d_in[13];
    const float* mb1    = (const float*)d_in[14];
    const float* mW2    = (const float*)d_in[15];
    const float* mb2    = (const float*)d_in[16];
    const float* lr_sc  = (const float*)d_in[17];
    const float* fg_sc  = (const float*)d_in[18];
    const float* Wg     = (const float*)d_in[19];
    const float* bg     = (const float*)d_in[20];

    char* ws = (char*)d_ws;
    size_t off = 0;
    auto alloc = [&](size_t bytes)->void*{
        void* p = ws + off; off += (bytes + 255) & ~(size_t)255; return p;
    };

    _Float16* xn_h  = (_Float16*)alloc((size_t)MM*DIMM*2);
    _Float16* wk_t  = (_Float16*)alloc((size_t)DIMM*DIMM*2);
    _Float16* wv_t  = (_Float16*)alloc((size_t)DIMM*DIMM*2);
    _Float16* wq_t  = (_Float16*)alloc((size_t)DIMM*DIMM*2);
    _Float16* wg_t  = (_Float16*)alloc((size_t)DIMM*DIMM*2);
    _Float16* wo_t  = (_Float16*)alloc((size_t)DIMM*DIMM*2);
    float* kraw = (float*)alloc((size_t)MM*DIMM*4);
    float* vraw = (float*)alloc((size_t)MM*DIMM*4);
    float* qraw = (float*)alloc((size_t)MM*DIMM*4);
    float* graw = (float*)alloc((size_t)MM*DIMM*4);
    float* kn   = (float*)alloc((size_t)MM*DIMM*4);
    float* vn   = (float*)alloc((size_t)MM*DIMM*4);
    float* qn   = (float*)alloc((size_t)MM*DIMM*4);
    _Float16* outs_h = (_Float16*)alloc((size_t)MM*DIMM*2);

    float* out_main   = (float*)d_out;                       // [B,S,DIM]
    float* out_states = out_main + (size_t)MM*DIMM;          // [L,B,H,D,D]

    // 1) input LN -> f16 activations
    ln_in_kernel<<<MM, 256, 0, stream>>>(x, ln_in_w, ln_in_b, xn_h);

    // 2) weight convert+transpose -> f16 [N,K]
    int wblocks = DIMM*DIMM/256;
    wconv_kernel<<<wblocks, 256, 0, stream>>>(Wk, wk_t);
    wconv_kernel<<<wblocks, 256, 0, stream>>>(Wv, wv_t);
    wconv_kernel<<<wblocks, 256, 0, stream>>>(Wq, wq_t);
    wconv_kernel<<<wblocks, 256, 0, stream>>>(Wg, wg_t);
    wconv_kernel<<<wblocks, 256, 0, stream>>>(Wo, wo_t);

    // 3) WMMA projection GEMMs (4096x768x768), 2x4 wave blocking
    gemm16_kernel<<<GEMM_BLOCKS, 256, 0, stream>>>(xn_h, wk_t, kraw);
    gemm16_kernel<<<GEMM_BLOCKS, 256, 0, stream>>>(xn_h, wv_t, vraw);
    gemm16_kernel<<<GEMM_BLOCKS, 256, 0, stream>>>(xn_h, wq_t, qraw);
    gemm16_kernel<<<GEMM_BLOCKS, 256, 0, stream>>>(xn_h, wg_t, graw);

    // 4) per-head LN + permute to time-major
    headln_kernel<<<3*BB*SS*HH, 64, 0, stream>>>(kraw, vraw, qraw, kn, vn, qn,
                                                 ln_k_w, ln_k_b, ln_v_w, ln_v_b,
                                                 ln_q_w, ln_q_b);

    // 5) sequential scan: one block per (b,h)
    scan_kernel<<<BB*HH, 256, 0, stream>>>(kn, vn, qn, mW1, mb1, mW2, mb2,
                                           lr_sc, fg_sc, outs_h, out_states);

    // 6) output WMMA GEMM with fused sigmoid gate
    gemm_out_kernel<<<GEMM_BLOCKS, 256, 0, stream>>>(outs_h, wo_t, graw, bg, out_main);
}
